// MambaLayer_9216999817338
// MI455X (gfx1250) — compile-verified, matrix operand order checked by reading, not run on hardware
//
#include <hip/hip_runtime.h>
#include <hip/hip_bf16.h>

// ---------------------------------------------------------------------------
// Vision-Mamba layer for MI455X (gfx1250, wave32, WMMA + TDM).
//
//   K_ln      : xn = LayerNorm(x_flat + pe)            -> f16 [16384,256]
//   K_in_proj : xz = xn @ in_proj_w^T  (WMMA f16,
//               weights staged to LDS via TENSOR_LOAD_TO_LDS, 8 waves/blk,
//               B fragments read back with ds_load_b128)
//   K_conv    : u  = silu(causal_conv4(xi)+b)          -> f16 [16384,512]
//   K_x_proj  : dbl = u @ x_proj_w^T   (WMMA f16)      -> f32 [16384,48]
//   K_scan    : fused dt_proj+softplus+selective scan
//               + u*D + silu(z) gating                 -> f16 [16384,512]
//   K_out_proj: out = y @ out_proj_w^T (WMMA f16)      -> f32 (B,256,H,W)
//
// ~15 GFLOP vs ~150 MB traffic -> memory bound at 23.3 TB/s; WMMA keeps the
// VALU free for the transcendental-heavy scan, f16 halves intermediate
// traffic, and the TDM stage removes 8x redundant weight fetches in in_proj.
// ---------------------------------------------------------------------------

#define DIMC   256
#define DSTATE 16
#define DCONV  4
#define DINNER 512
#define DTRANK 16
#define NB     4
#define LSEQ   4096      // 64*64
#define NTOK   (NB*LSEQ) // 16384

typedef _Float16     v16h __attribute__((ext_vector_type(16)));
typedef _Float16     v8h  __attribute__((ext_vector_type(8)));
typedef float        v8f  __attribute__((ext_vector_type(8)));
typedef unsigned int v4u  __attribute__((ext_vector_type(4)));
typedef int          v8i  __attribute__((ext_vector_type(8)));
typedef int          v4i  __attribute__((ext_vector_type(4)));

// ---- WMMA helpers ---------------------------------------------------------

__device__ __forceinline__ v8f wmma16(v16h a, v16h b, v8f c) {
  // D = A(16x32 f16) x B(32x16 f16) + C(16x16 f32)
  return __builtin_amdgcn_wmma_f32_16x16x32_f16(
      /*neg_a=*/false, a, /*neg_b=*/false, b,
      /*c_mod=*/(short)0, c, /*reuse_a=*/false, /*reuse_b=*/false);
}

// 16x32 f16 fragment, rows row-major with leading dim `ld` (elements).
// ISA layout (16-bit A 16x32): lanes 0-15 hold row M=lane, K=k0+[0..7] and
// k0+[16..23]; lanes 16-31 hold row M=lane-16, K=k0+[8..15] and k0+[24..31].
// Same addressing serves the B fragment for [out,in]-major weights.
__device__ __forceinline__ v16h load_frag(const _Float16* __restrict__ base,
                                          int ld, int row0, int k0) {
  const int lane = threadIdx.x & 31;
  const _Float16* p =
      base + (size_t)(row0 + (lane & 15)) * ld + k0 + ((lane >> 4) << 3);
  v8h lo = *(const v8h*)(p);
  v8h hi = *(const v8h*)(p + 16);
  return __builtin_shufflevector(lo, hi, 0, 1, 2, 3, 4, 5, 6, 7,
                                 8, 9, 10, 11, 12, 13, 14, 15);
}

#define CAT16(lo, hi)                                              \
  __builtin_shufflevector(lo, hi, 0, 1, 2, 3, 4, 5, 6, 7, 8, 9, 10, 11, 12, \
                          13, 14, 15)

// Four 16x32 B fragments (rows f*16..f*16+15 of the 64x256 LDS weight tile)
// read via explicit ds_load_b128.  The tile is written by the Tensor Data
// Mover, which the compiler cannot see as a store, so the reads must be
// inline asm (plain loads of a "never-stored" LDS object fold to undef).
// abyte = 2*((lane&15)*256 + ((lane>>4)<<3) + k); fragment f at +f*8192,
// hi half at +32.  One s_wait_dscnt for all eight loads.
struct BFrags { v16h b0, b1, b2, b3; };

__device__ __forceinline__ BFrags load_bfrags_lds(unsigned abyte) {
  v8h b0l, b0h, b1l, b1h, b2l, b2h, b3l, b3h;
  asm volatile(
      "ds_load_b128 %0, %8\n\t"
      "ds_load_b128 %1, %8 offset:32\n\t"
      "ds_load_b128 %2, %8 offset:8192\n\t"
      "ds_load_b128 %3, %8 offset:8224\n\t"
      "ds_load_b128 %4, %8 offset:16384\n\t"
      "ds_load_b128 %5, %8 offset:16416\n\t"
      "ds_load_b128 %6, %8 offset:24576\n\t"
      "ds_load_b128 %7, %8 offset:24608\n\t"
      "s_wait_dscnt 0x0"
      : "=&v"(b0l), "=&v"(b0h), "=&v"(b1l), "=&v"(b1h),
        "=&v"(b2l), "=&v"(b2h), "=&v"(b3l), "=&v"(b3h)
      : "v"(abyte));
  BFrags r;
  r.b0 = CAT16(b0l, b0h);
  r.b1 = CAT16(b1l, b1h);
  r.b2 = CAT16(b2l, b2h);
  r.b3 = CAT16(b3l, b3h);
  return r;
}

// ---- small utility kernels ------------------------------------------------

__global__ void __launch_bounds__(256)
k_f32_to_f16(const float* __restrict__ src, _Float16* __restrict__ dst, int n) {
  int i = blockIdx.x * 256 + threadIdx.x;
  if (i < n) dst[i] = (_Float16)src[i];
}

// One block (256 threads) per token: xn = LN(x[b,:,l] + pe[l,:]) -> f16
__global__ void __launch_bounds__(256)
k_ln(const float* __restrict__ x, const float* __restrict__ pe,
     const float* __restrict__ w, const float* __restrict__ bwt,
     _Float16* __restrict__ xn) {
  const int row = blockIdx.x;          // b*L + l
  const int c   = threadIdx.x;         // channel
  const int b   = row >> 12;
  const int l   = row & (LSEQ - 1);
  float v = x[((size_t)b * DIMC + c) * LSEQ + l] + pe[(size_t)l * DIMC + c];

  __shared__ float s[DIMC];
  s[c] = v;
  __syncthreads();
  for (int off = 128; off > 0; off >>= 1) {
    if (c < off) s[c] += s[c + off];
    __syncthreads();
  }
  const float mu = s[0] * (1.0f / DIMC);
  __syncthreads();
  const float d = v - mu;
  s[c] = d * d;
  __syncthreads();
  for (int off = 128; off > 0; off >>= 1) {
    if (c < off) s[c] += s[c + off];
    __syncthreads();
  }
  const float var = s[0] * (1.0f / DIMC);
  const float r = rsqrtf(var + 1e-5f);
  xn[(size_t)row * DIMC + c] = (_Float16)(d * r * w[c] + bwt[c]);
}

// ---- in_proj GEMM with TDM-staged weights ---------------------------------
// xz[m, n] = sum_k xn[m,k] * Win[n,k]   M=16384 N=1024 K=256
// Block: 8 waves share one 64-wide N tile; weight tile (64x256 f16 = 32 KB)
// is DMA'd to LDS once by the Tensor Data Mover, then each wave computes a
// 16x64 output tile with B fragments from LDS and A fragments from global.

__global__ void __launch_bounds__(256)
k_in_proj(const _Float16* __restrict__ xn, const _Float16* __restrict__ w,
          float* __restrict__ xz) {
  __shared__ _Float16 wtile[64 * DIMC];  // 32 KB, LDS offset 0
  (void)wtile;                           // written by TDM, read via asm

  const int n0   = blockIdx.x * 64;
  const int wave = threadIdx.x >> 5;
  const int m0   = (blockIdx.y * 8 + wave) * 16;

  if (threadIdx.x < 32) {
    // Tensor DMA descriptor (2-D): tensor = Win [1024 x 256] f16, tile =
    // rows [n0, n0+64) x cols [0, 256), dest LDS offset 0.
    const unsigned long long ga =
        (unsigned long long)(uintptr_t)w + (unsigned long long)n0 * DIMC * 2u;
    v4u g0;
    g0[0] = 1u;                                   // count=1, user mode
    g0[1] = 0u;                                   // lds_addr = 0 (wtile)
    g0[2] = (unsigned)(ga & 0xffffffffu);         // global_addr[31:0]
    g0[3] = (unsigned)((ga >> 32) & 0x01ffffffu)  // global_addr[56:32]
            | (2u << 30);                         // type = 2 ("image")
    v8i g1;
    g1[0] = 0x00010000;          // wg_mask=0, data_size=1 (2 bytes)
    g1[1] = (int)(DIMC << 16);   // tensor_dim0[15:0]=256 in bits[63:48]
    g1[2] = (int)(1024u << 16);  // td0[31:16]=0 | tensor_dim1[15:0]=1024
    g1[3] = (int)(DIMC << 16);   // td1[31:16]=0 | tile_dim0=256
    g1[4] = 64;                  // tile_dim1=64 rows, tile_dim2=0
    g1[5] = DIMC;                // tensor_dim0_stride = 256 elements
    g1[6] = 0;
    g1[7] = 0;
    v4i gz4 = {0, 0, 0, 0};      // groups 2/3 unused for 2-D tile
    v8i gz8 = {0, 0, 0, 0, 0, 0, 0, 0};
    // 6-arg toolchain signature: (g0, g1, g2, g3, extra_group, cpol)
    __builtin_amdgcn_tensor_load_to_lds(g0, g1, gz4, gz4, gz8, 0);
    __builtin_amdgcn_s_wait_tensorcnt(0);
  }
  __syncthreads();

  const int lane = threadIdx.x & 31;
  // Byte address inside the LDS tile for this lane's fragment slice.
  const unsigned abase =
      (unsigned)(((lane & 15) * DIMC + ((lane >> 4) << 3)) * 2);

  v8f c0 = {}, c1 = {}, c2 = {}, c3 = {};
  for (int k = 0; k < DIMC; k += 32) {
    v16h a = load_frag(xn, DIMC, m0, k);
    BFrags bf = load_bfrags_lds(abase + (unsigned)(k * 2));
    c0 = wmma16(a, bf.b0, c0);
    c1 = wmma16(a, bf.b1, c1);
    c2 = wmma16(a, bf.b2, c2);
    c3 = wmma16(a, bf.b3, c3);
  }
  const int col = lane & 15;
  const int mo  = (lane >> 4) << 3;
#pragma unroll
  for (int v = 0; v < 8; ++v) {
    const size_t m = (size_t)(m0 + v + mo);
    xz[m * 1024 + n0 +  0 + col] = c0[v];
    xz[m * 1024 + n0 + 16 + col] = c1[v];
    xz[m * 1024 + n0 + 32 + col] = c2[v];
    xz[m * 1024 + n0 + 48 + col] = c3[v];
  }
}

// u = silu(depthwise causal conv over xi (= xz[:, :512])) -> f16
__global__ void __launch_bounds__(256)
k_conv(const float* __restrict__ xz, const float* __restrict__ cw,
       const float* __restrict__ cb, _Float16* __restrict__ uh) {
  const int idx = blockIdx.x * 256 + threadIdx.x;  // NTOK*512 threads
  const int d   = idx & (DINNER - 1);
  const int row = idx >> 9;
  const int b   = row >> 12;
  const int l   = row & (LSEQ - 1);
  float acc = cb[d];
#pragma unroll
  for (int j = 0; j < DCONV; ++j) {
    const int ll = l - (DCONV - 1) + j;
    if (ll >= 0)
      acc += xz[((size_t)(b << 12) + ll) * 1024 + d] * cw[d * DCONV + j];
  }
  const float u = acc / (1.0f + __expf(-acc));  // silu
  uh[(size_t)row * DINNER + d] = (_Float16)u;
}

// dbl[m, e] = sum_k u[m,k] * Wxp[e,k]   M=16384 N=48 K=512, tile 16x48
__global__ void __launch_bounds__(32)
k_x_proj(const _Float16* __restrict__ uh, const _Float16* __restrict__ w,
         float* __restrict__ dbl) {
  const int m0 = blockIdx.x * 16;
  v8f c0 = {}, c1 = {}, c2 = {};
  for (int k = 0; k < DINNER; k += 32) {
    v16h a  = load_frag(uh, DINNER, m0, k);
    v16h b0 = load_frag(w, DINNER,  0, k);
    v16h b1 = load_frag(w, DINNER, 16, k);
    v16h b2 = load_frag(w, DINNER, 32, k);
    c0 = wmma16(a, b0, c0);
    c1 = wmma16(a, b1, c1);
    c2 = wmma16(a, b2, c2);
  }
  const int lane = threadIdx.x & 31;
  const int col  = lane & 15;
  const int mo   = (lane >> 4) << 3;
#pragma unroll
  for (int v = 0; v < 8; ++v) {
    const size_t m = (size_t)(m0 + v + mo);
    dbl[m * 48 +  0 + col] = c0[v];
    dbl[m * 48 + 16 + col] = c1[v];
    dbl[m * 48 + 32 + col] = c2[v];
  }
}

// Fused: dt_proj (rank 16) + softplus + selective scan + u*D + silu(z) gate.
// grid = (2 d-chunks, 4 batches), block = 256 threads (one channel each).
__global__ void __launch_bounds__(256)
k_scan(const float* __restrict__ dbl, const _Float16* __restrict__ uh,
       const float* __restrict__ xz, const float* __restrict__ dtw,
       const float* __restrict__ dtb, const float* __restrict__ A_log,
       const float* __restrict__ Dp, _Float16* __restrict__ yh) {
  const int d = blockIdx.x * 256 + threadIdx.x;  // 0..511
  const int b = blockIdx.y;
  float A[DSTATE], wr[DTRANK], h[DSTATE];
#pragma unroll
  for (int n = 0; n < DSTATE; ++n) {
    A[n]  = -__expf(A_log[(size_t)d * DSTATE + n]);
    wr[n] = dtw[(size_t)d * DTRANK + n];
    h[n]  = 0.0f;
  }
  const float bias = dtb[d];
  const float Dd   = Dp[d];
  __shared__ float sh[48];  // [dt(16) | B(16) | C(16)] for current step

  for (int l = 0; l < LSEQ; ++l) {
    const size_t row = (size_t)(b << 12) + l;
    if (threadIdx.x < 48) sh[threadIdx.x] = dbl[row * 48 + threadIdx.x];
    __syncthreads();

    float dt = bias;
#pragma unroll
    for (int r = 0; r < DTRANK; ++r) dt += sh[r] * wr[r];
    dt = (dt > 20.0f) ? dt : logf(1.0f + __expf(dt));  // softplus

    const float ut = (float)uh[row * DINNER + d];
    const float du = dt * ut;
    float acc = 0.0f;
#pragma unroll
    for (int n = 0; n < DSTATE; ++n) {
      const float e = __expf(dt * A[n]);
      h[n] = e * h[n] + du * sh[16 + n];
      acc += h[n] * sh[32 + n];
    }
    float y = acc + ut * Dd;
    const float z = xz[row * 1024 + DINNER + d];
    y *= z / (1.0f + __expf(-z));  // * silu(z)
    yh[row * DINNER + d] = (_Float16)y;
    __syncthreads();
  }
}

// out[b, c, l] = sum_k y[m,k] * Wout[c,k]  M=16384 N=256 K=512, tile 16x64
__global__ void __launch_bounds__(32)
k_out_proj(const _Float16* __restrict__ yh, const _Float16* __restrict__ w,
           float* __restrict__ out) {
  const int n0 = blockIdx.x * 64;
  const int m0 = blockIdx.y * 16;
  v8f c0 = {}, c1 = {}, c2 = {}, c3 = {};
  for (int k = 0; k < DINNER; k += 32) {
    v16h a  = load_frag(yh, DINNER, m0, k);
    v16h b0 = load_frag(w, DINNER, n0 +  0, k);
    v16h b1 = load_frag(w, DINNER, n0 + 16, k);
    v16h b2 = load_frag(w, DINNER, n0 + 32, k);
    v16h b3 = load_frag(w, DINNER, n0 + 48, k);
    c0 = wmma16(a, b0, c0);
    c1 = wmma16(a, b1, c1);
    c2 = wmma16(a, b2, c2);
    c3 = wmma16(a, b3, c3);
  }
  const int lane = threadIdx.x & 31;
  const int col  = lane & 15;
  const int mo   = (lane >> 4) << 3;
#pragma unroll
  for (int v = 0; v < 8; ++v) {
    const int m  = m0 + v + mo;       // token row
    const int bb = m >> 12;
    const int l  = m & (LSEQ - 1);
    out[((size_t)bb * DIMC + n0 +  0 + col) * LSEQ + l] = c0[v];
    out[((size_t)bb * DIMC + n0 + 16 + col) * LSEQ + l] = c1[v];
    out[((size_t)bb * DIMC + n0 + 32 + col) * LSEQ + l] = c2[v];
    out[((size_t)bb * DIMC + n0 + 48 + col) * LSEQ + l] = c3[v];
  }
}

// ---------------------------------------------------------------------------

extern "C" void kernel_launch(void* const* d_in, const int* in_sizes, int n_in,
                              void* d_out, int out_size, void* d_ws,
                              size_t ws_size, hipStream_t stream) {
  const float* x          = (const float*)d_in[0];
  const float* pe         = (const float*)d_in[1];
  /* d_in[2] = order (identity, unused by reference math) */
  const float* norm_w     = (const float*)d_in[3];
  const float* norm_b     = (const float*)d_in[4];
  const float* in_proj_w  = (const float*)d_in[5];
  const float* conv_w     = (const float*)d_in[6];
  const float* conv_b     = (const float*)d_in[7];
  const float* x_proj_w   = (const float*)d_in[8];
  const float* dt_proj_w  = (const float*)d_in[9];
  const float* dt_proj_b  = (const float*)d_in[10];
  const float* A_log      = (const float*)d_in[11];
  const float* Dp         = (const float*)d_in[12];
  const float* out_proj_w = (const float*)d_in[13];
  float* out = (float*)d_out;

  char* ws = (char*)d_ws;
  size_t off = 0;
  auto carve = [&](size_t bytes) -> void* {
    void* p = ws + off;
    off = (off + bytes + 255) & ~(size_t)255;
    return p;
  };
  _Float16* xn_h   = (_Float16*)carve((size_t)NTOK * DIMC * 2);     //  8 MB
  _Float16* wh_in  = (_Float16*)carve((size_t)1024 * DIMC * 2);     // .5 MB
  _Float16* wh_xp  = (_Float16*)carve((size_t)48 * DINNER * 2);     // 48 KB
  _Float16* wh_out = (_Float16*)carve((size_t)DIMC * DINNER * 2);   // .25MB
  float*    xz     = (float*)carve((size_t)NTOK * 1024 * 4);        // 64 MB
  _Float16* uh     = (_Float16*)carve((size_t)NTOK * DINNER * 2);   // 16 MB
  float*    dbl    = (float*)carve((size_t)NTOK * 48 * 4);          //  3 MB
  _Float16* yh     = (_Float16*)carve((size_t)NTOK * DINNER * 2);   // 16 MB

  // weights -> f16 (tiny; done per launch, deterministic)
  k_f32_to_f16<<<(1024 * DIMC + 255) / 256, 256, 0, stream>>>(in_proj_w, wh_in,
                                                              1024 * DIMC);
  k_f32_to_f16<<<(48 * DINNER + 255) / 256, 256, 0, stream>>>(x_proj_w, wh_xp,
                                                              48 * DINNER);
  k_f32_to_f16<<<(DIMC * DINNER + 255) / 256, 256, 0, stream>>>(
      out_proj_w, wh_out, DIMC * DINNER);

  // 1) LayerNorm(x_flat + pe)
  k_ln<<<NTOK, 256, 0, stream>>>(x, pe, norm_w, norm_b, xn_h);

  // 2) in_proj GEMM (WMMA + TDM weight staging): 8 waves/block, 128 rows/blk
  k_in_proj<<<dim3(1024 / 64, NTOK / 128), 256, 0, stream>>>(xn_h, wh_in, xz);

  // 3) causal depthwise conv + silu
  k_conv<<<(NTOK * DINNER) / 256, 256, 0, stream>>>(xz, conv_w, conv_b, uh);

  // 4) x_proj GEMM (WMMA): 16384x48 = u @ W^T
  k_x_proj<<<NTOK / 16, 32, 0, stream>>>(uh, wh_xp, dbl);

  // 5) fused dt_proj + softplus + selective scan + gating
  k_scan<<<dim3(DINNER / 256, NB), 256, 0, stream>>>(dbl, uh, xz, dt_proj_w,
                                                     dt_proj_b, A_log, Dp, yh);

  // 6) out_proj GEMM (WMMA), scattered straight into (B, C, H, W) layout
  k_out_proj<<<dim3(DIMC / 64, NTOK / 16), 32, 0, stream>>>(yh, wh_out, out);
}
